// HandwritingSynthesisModel_7490422964673
// MI455X (gfx1250) — matrix operand order repaired and checked by reading, not compiled, not run on hardware
//
#include <hip/hip_runtime.h>
#include <hip/hip_bf16.h>

// ---------------------------------------------------------------------------
// Handwriting synthesis (Graves attention, 3x LSTM H=256, B=128, T=800)
// Persistent single-workgroup recurrence on one WGP using CDNA5 bf16 WMMA.
// Round 2: 32-bit weight offsets (SGPR base + voffset addressing, no spills),
// segmented K loop (no per-kt branch), vectorized attention GEMV / h copies.
// ---------------------------------------------------------------------------

typedef __attribute__((ext_vector_type(16))) __bf16 v16bf;
typedef __attribute__((ext_vector_type(8)))  float  v8f;
typedef __attribute__((ext_vector_type(4)))  int    v4i;

union Frag { v16bf v; v4i i[2]; };

__device__ __forceinline__ v8f wmma_bf16(const Frag& a, const Frag& b, v8f c) {
  return __builtin_amdgcn_wmma_f32_16x16x32_bf16(false, a.v, false, b.v, (short)0, c,
                                                 false, false);
}

__device__ __forceinline__ unsigned short f2bf(float f) {
  unsigned u = __float_as_uint(f);
  u += 0x7FFFu + ((u >> 16) & 1u);          // round-to-nearest-even
  return (unsigned short)(u >> 16);
}
__device__ __forceinline__ float bf2f(unsigned short h) {
  return __uint_as_float(((unsigned)h) << 16);
}
__device__ __forceinline__ float bflo(int u) { return __uint_as_float(((unsigned)u) << 16); }
__device__ __forceinline__ float bfhi(int u) { return __uint_as_float(((unsigned)u) & 0xFFFF0000u); }
__device__ __forceinline__ float sigf(float x) { return 1.f / (1.f + __expf(-x)); }

// ---------------- sizes / layout constants ----------------
#define TSTEPS 800
#define BATCH  128
#define HDIM   256
#define NCOLS  1024          // 4 gates * H, gate-interleaved: n = 4*j + g
#define KT1    11            // K tiles LSTM1: [x,1,pad | w | h1]      -> K=352
#define KT23   19            // K tiles LSTM2/3: [x,1,pad | w | ha|hb] -> K=608
#define LSTM_SW_ELEMS(KT) ((KT) * NCOLS * 32)   // bf16 elements

// LDS offsets (bytes)
#define OFF_H1   0
#define OFF_H2   65536
#define OFF_H3   131072
#define OFF_HXW  196608      // [128][96] bf16: cols 0..31 = x,1,pad ; 32..95 = w
#define OFF_KAP  221184      // [128][10] f32
#define OFF_WIN  226304      // [128][32] f32 (alpha 0..9, beta 10..19, k 20..29)
#define OFF_PHI  242688      // [128][64] f32
#define OFF_ZPR  226304      // [128][128] f32, aliases WIN/PHI (disjoint in time)
#define SMEM_BYTES 291840

// workspace byte offsets
#define WS_W1  0u
#define WS_W2  720896u
#define WS_W3  1966080u
#define WS_WP  3211264u
#define WS_C1  3276800u
#define WS_C2  3407872u
#define WS_C3  3538944u

// ---------------------------------------------------------------------------
// Weight prep: fp32 -> bf16, gate-interleaved columns, K-concatenated rows,
// stored in the exact 32B-per-(tile,col,khalf) swizzle the B-fragment loads.
// K map: tile0 = [x0,x1,x2,1,pad*28]; tiles1-2 = w(64); tiles3-10 = ha(256);
//        tiles11-18 = hb(256).
// ---------------------------------------------------------------------------
__device__ __forceinline__ void emit_lstm(unsigned short* dst, int e,
    const float* Wx, int sx, const float* bias,
    const float* Wha, int sha, int oha, const float* Whb)
{
  int q  = e & 31;
  int N  = (e >> 5) & 1023;
  int kt = e >> 15;
  int k  = kt * 32 + q;
  int o  = (N & 3) * 256 + (N >> 2);        // original gate-major column
  float v;
  if      (k < 3)   v = Wx[o * sx + k];
  else if (k == 3)  v = bias[o];
  else if (k < 32)  v = 0.f;
  else if (k < 96)  v = Wx[o * sx + 3 + (k - 32)];
  else if (k < 352) v = Wha[o * sha + oha + (k - 96)];
  else              v = Whb[o * 256 + (k - 352)];
  dst[e] = f2bf(v);
}

__global__ void hs_prep_weights(
    const float* __restrict__ W1_ih, const float* __restrict__ W1_hh, const float* __restrict__ b1,
    const float* __restrict__ W2_ih, const float* __restrict__ W2_hh, const float* __restrict__ b2,
    const float* __restrict__ W3_ih, const float* __restrict__ W3_hh, const float* __restrict__ b3,
    const float* __restrict__ We,  const float* __restrict__ Wpi,
    const float* __restrict__ Wm1, const float* __restrict__ Wm2,
    const float* __restrict__ Ws1, const float* __restrict__ Ws2, const float* __restrict__ Wr,
    unsigned short* __restrict__ Wsw1, unsigned short* __restrict__ Wsw2,
    unsigned short* __restrict__ Wsw3, unsigned short* __restrict__ Wswp)
{
  const int S1 = LSTM_SW_ELEMS(KT1);        // 360448
  const int S2 = LSTM_SW_ELEMS(KT23);       // 622592
  int i = blockIdx.x * blockDim.x + threadIdx.x;
  if (i >= S1 + 2 * S2 + 32768) return;
  if (i < S1) {
    emit_lstm(Wsw1, i, W1_ih, 67, b1, W1_hh, 256, 0, W1_hh);
  } else if (i < S1 + S2) {
    emit_lstm(Wsw2, i - S1, W2_ih, 323, b2, W2_ih, 323, 67, W2_hh);
  } else if (i < S1 + 2 * S2) {
    emit_lstm(Wsw3, i - S1 - S2, W3_ih, 323, b3, W3_ih, 323, 67, W3_hh);
  } else {
    int e  = i - S1 - 2 * S2;               // projection: [8 kt][128 N][32]
    int q  = e & 31;
    int N  = (e >> 5) & 127;
    int kt = e >> 12;
    int k  = kt * 32 + q;
    float v;
    if      (N == 0)   v = We[k];
    else if (N <= 20)  v = Wpi[(N - 1) * 256 + k];
    else if (N <= 40)  v = Wm1[(N - 21) * 256 + k];
    else if (N <= 60)  v = Wm2[(N - 41) * 256 + k];
    else if (N <= 80)  v = Ws1[(N - 61) * 256 + k];
    else if (N <= 100) v = Ws2[(N - 81) * 256 + k];
    else if (N <= 120) v = Wr[(N - 101) * 256 + k];
    else               v = 0.f;
    Wswp[e] = f2bf(v);
  }
}

// ---------------------------------------------------------------------------
// One straight-line K segment: nkt K-tiles of A from one LDS buffer, B from
// the swizzled bf16 weight stream (32-bit byte offsets off an SGPR base).
// ---------------------------------------------------------------------------
__device__ __forceinline__ void mm_segment(
    const unsigned short* __restrict__ abase, int astr, int ktbase, int nkt, int KT,
    const char* __restrict__ WswB, v8f acc[2][4],
    int lane, int m_pair, int n_oct, int pass)
{
  const int ml = lane & 15, hf = lane >> 4;
  const int row0 = (m_pair * 2) * 16 + ml;
  const unsigned ncol0 = (unsigned)((n_oct * 8 + pass * 4) * 16 + ml);
  for (int kk = 0; kk < nkt; ++kk) {
    const int kt = ktbase + kk;
    Frag af[2];
#pragma unroll
    for (int a = 0; a < 2; ++a) {
      const unsigned short* p = abase + (row0 + a * 16) * astr + kk * 32 + hf * 16;
      af[a].i[0] = *(const v4i*)p;          // K 0..7   (hf=0) / 16..23 (hf=1)
      af[a].i[1] = *(const v4i*)(p + 8);    // K 8..15  (hf=0) / 24..31 (hf=1)
    }
#pragma unroll
    for (int i = 0; i < 4; ++i) {
      unsigned boff = (((unsigned)kt * (unsigned)NCOLS + ncol0 + (unsigned)(i * 16)) * 2u
                       + (unsigned)hf) * 32u;
      Frag bf;
      bf.i[0] = *(const v4i*)(WswB + boff);
      bf.i[1] = *(const v4i*)(WswB + boff + 16);
      acc[0][i] = wmma_bf16(af[0], bf, acc[0][i]);
      acc[1][i] = wmma_bf16(af[1], bf, acc[1][i]);
    }
    if (kt + 1 < KT) {
      unsigned poff = (((unsigned)(kt + 1) * (unsigned)NCOLS + ncol0) * 2u
                       + (unsigned)hf) * 32u;
      __builtin_prefetch(WswB + poff, 0, 1);   // global_prefetch: next K tile
    }
  }
}

// ---------------------------------------------------------------------------
// One LSTM layer: z = A(bf16,LDS) x Wsw(bf16,L2) via WMMA, gate math via quad
// shuffles, h_new staged into Htmp (caller copies into the live h buffer).
// Each wave: 2 M-tiles x 8 N-tiles, in two 4-N passes (acc pressure ~64 VGPR).
// ---------------------------------------------------------------------------
__device__ __forceinline__ void lstm_phase(
    int KT,
    const unsigned short* __restrict__ Hxw,
    const unsigned short* __restrict__ Ha,
    const unsigned short* __restrict__ Hb,
    const char* __restrict__ WswB,
    float* __restrict__ cbuf,
    unsigned short* __restrict__ Htmp,
    int lane, int m_pair, int n_oct)
{
  const int hf  = lane >> 4;   // accumulator row half (rows 8*hf..8*hf+7)
  const int gq  = lane & 3;    // gate id within quad
  const int jq  = (lane >> 2) & 3;
  const v8f z8 = {0.f, 0.f, 0.f, 0.f, 0.f, 0.f, 0.f, 0.f};

  for (int pass = 0; pass < 2; ++pass) {
    v8f acc[2][4];
#pragma unroll
    for (int a = 0; a < 2; ++a)
#pragma unroll
      for (int i = 0; i < 4; ++i) acc[a][i] = z8;

    mm_segment(Hxw, 96, 0, 3, KT, WswB, acc, lane, m_pair, n_oct, pass);
    mm_segment(Ha, 256, 3, 8, KT, WswB, acc, lane, m_pair, n_oct, pass);
    if (KT > 11)
      mm_segment(Hb, 256, 11, 8, KT, WswB, acc, lane, m_pair, n_oct, pass);

    // gate math: quad (4 lanes) holds i,f,g,o for one hidden dim j
#pragma unroll
    for (int a = 0; a < 2; ++a)
#pragma unroll
      for (int i = 0; i < 4; ++i) {
        int n_t = n_oct * 8 + pass * 4 + i;
#pragma unroll
        for (int v = 0; v < 8; ++v) {
          float z  = acc[a][i][v];
          int   qb = lane & ~3;
          float zi = __shfl(z, qb + 0, 32);
          float zf = __shfl(z, qb + 1, 32);
          float zg = __shfl(z, qb + 2, 32);
          float zo = __shfl(z, qb + 3, 32);
          if (gq == 0) {
            int row = (m_pair * 2 + a) * 16 + v + 8 * hf;
            int j   = n_t * 4 + jq;
            float cold = cbuf[row * 256 + j];
            float cn   = sigf(zf) * cold + sigf(zi) * tanhf(zg);
            float hn   = sigf(zo) * tanhf(cn);
            cbuf[row * 256 + j] = cn;
            Htmp[row * 256 + j] = f2bf(hn);
          }
        }
      }
  }
}

// ---------------------------------------------------------------------------
// Persistent recurrence kernel: one workgroup, 32 waves, 800 sequential steps.
// ---------------------------------------------------------------------------
__global__ __launch_bounds__(1024) void hs_recurrence(
    const float* __restrict__ x, const float* __restrict__ cmat,
    const float* __restrict__ Ww,  const float* __restrict__ bw,
    const float* __restrict__ be,  const float* __restrict__ bpi,
    const float* __restrict__ bm1, const float* __restrict__ bm2,
    const float* __restrict__ bs1, const float* __restrict__ bs2,
    const float* __restrict__ br,
    const char* __restrict__ Wsw1, const char* __restrict__ Wsw2,
    const char* __restrict__ Wsw3, const char* __restrict__ Wswp,
    float* __restrict__ c1g, float* __restrict__ c2g, float* __restrict__ c3g,
    float* __restrict__ out)
{
  extern __shared__ char smem[];
  unsigned short* H1   = (unsigned short*)(smem + OFF_H1);
  unsigned short* H2   = (unsigned short*)(smem + OFF_H2);
  unsigned short* H3   = (unsigned short*)(smem + OFF_H3);
  unsigned short* Hxw  = (unsigned short*)(smem + OFF_HXW);
  float*          kap  = (float*)(smem + OFF_KAP);
  float*          win  = (float*)(smem + OFF_WIN);
  float*          phi  = (float*)(smem + OFF_PHI);
  float*          zpr  = (float*)(smem + OFF_ZPR);
  unsigned short* Htmp = (unsigned short*)(smem + OFF_ZPR);   // time-aliased

  const int tid    = threadIdx.x;
  const int lane   = tid & 31;
  const int wave   = tid >> 5;
  const int m_pair = wave & 3;
  const int n_oct  = wave >> 2;
  const int ml     = lane & 15;
  const int hf     = lane >> 4;
  const unsigned short ONEBF = 0x3F80;

  // ---- carry init: h=0, c=0, w=1, kappa=0 ----
  for (int i = tid; i < BATCH * HDIM; i += 1024) {
    H1[i] = 0; H2[i] = 0; H3[i] = 0;
    c1g[i] = 0.f; c2g[i] = 0.f; c3g[i] = 0.f;
  }
  for (int i = tid; i < BATCH * 96; i += 1024) Hxw[i] = ((i % 96) >= 32) ? ONEBF : 0;
  for (int i = tid; i < BATCH * 10; i += 1024) kap[i] = 0.f;
  __syncthreads();

  const size_t TBo = (size_t)TSTEPS * BATCH;        // 102400
  const size_t NGo = TBo * 20;                      // 2048000

  for (int t = 0; t < TSTEPS; ++t) {
    // ---- stage x_t (and the bias-"1" input) ----
    for (int i = tid; i < BATCH * 32; i += 1024) {
      int b = i >> 5, col = i & 31;
      float v = (col < 3) ? x[((size_t)t * BATCH + b) * 3 + col] : (col == 3 ? 1.f : 0.f);
      Hxw[b * 96 + col] = f2bf(v);
    }
    __syncthreads();

    // ================= LSTM 1 =================
    lstm_phase(KT1, Hxw, H1, H1, Wsw1, c1g, Htmp, lane, m_pair, n_oct);
    __syncthreads();
    for (int i = tid; i < 4096; i += 1024) ((v4i*)H1)[i] = ((const v4i*)Htmp)[i];
    __syncthreads();

    // ================= attention window =================
    for (int idx = tid; idx < BATCH * 30; idx += 1024) {
      int b = idx / 30, r = idx % 30;
      const unsigned short* hb = H1 + b * 256;
      const float* wr = Ww + r * 256;
      float s = bw[r];
      for (int k = 0; k < 256; k += 8) {
        v4i hv = *(const v4i*)(hb + k);
        const float* w8 = wr + k;
        s += bflo(hv.x) * w8[0] + bfhi(hv.x) * w8[1]
           + bflo(hv.y) * w8[2] + bfhi(hv.y) * w8[3]
           + bflo(hv.z) * w8[4] + bfhi(hv.z) * w8[5]
           + bflo(hv.w) * w8[6] + bfhi(hv.w) * w8[7];
      }
      win[b * 32 + r] = __expf(s);
    }
    __syncthreads();
    for (int idx = tid; idx < BATCH * 10; idx += 1024) {
      int b = idx / 10, kk = idx % 10;
      kap[idx] += 0.1f * win[b * 32 + 20 + kk];
    }
    __syncthreads();
    for (int idx = tid; idx < BATCH * 64; idx += 1024) {
      int b = idx >> 6, u = idx & 63;
      float s = 0.f;
      for (int k = 0; k < 10; ++k) {
        float d = kap[b * 10 + k] - (float)u;
        s += win[b * 32 + k] * __expf(-win[b * 32 + 10 + k] * d * d);
      }
      phi[idx] = s;
    }
    __syncthreads();
    for (int idx = tid; idx < BATCH * 64; idx += 1024) {
      int b = idx >> 6, a = idx & 63;
      const float* cb = cmat + (size_t)b * 4096 + a;
      const float* pb = phi + b * 64;
      float s = 0.f;
      for (int u = 0; u < 64; ++u) s += pb[u] * cb[u * 64];
      Hxw[b * 96 + 32 + a] = f2bf(s);
    }
    __syncthreads();

    // ================= LSTM 2 / LSTM 3 =================
    lstm_phase(KT23, Hxw, H1, H2, Wsw2, c2g, Htmp, lane, m_pair, n_oct);
    __syncthreads();
    for (int i = tid; i < 4096; i += 1024) ((v4i*)H2)[i] = ((const v4i*)Htmp)[i];
    __syncthreads();

    lstm_phase(KT23, Hxw, H2, H3, Wsw3, c3g, Htmp, lane, m_pair, n_oct);
    __syncthreads();
    for (int i = tid; i < 4096; i += 1024) ((v4i*)H3)[i] = ((const v4i*)Htmp)[i];
    __syncthreads();

    // ================= output projection (h3 x [256,128]) =================
    {
      const int nt  = wave >> 2;               // 0..7
      const int mt0 = (2 * wave) & 7;          // wave owns tiles (mt0,nt),(mt0+1,nt)
      const v8f z8 = {0.f, 0.f, 0.f, 0.f, 0.f, 0.f, 0.f, 0.f};
      v8f p0 = z8, p1 = z8;
      const unsigned ncp = (unsigned)(nt * 16 + ml);
      for (int kt = 0; kt < 8; ++kt) {
        int r0 = mt0 * 16 + ml;
        const unsigned short* pa0 = H3 + r0 * 256 + kt * 32 + hf * 16;
        const unsigned short* pa1 = H3 + (r0 + 16) * 256 + kt * 32 + hf * 16;
        Frag a0, a1, bf;
        a0.i[0] = *(const v4i*)pa0; a0.i[1] = *(const v4i*)(pa0 + 8);
        a1.i[0] = *(const v4i*)pa1; a1.i[1] = *(const v4i*)(pa1 + 8);
        unsigned boff = (((unsigned)kt * 128u + ncp) * 2u + (unsigned)hf) * 32u;
        bf.i[0] = *(const v4i*)(Wswp + boff);
        bf.i[1] = *(const v4i*)(Wswp + boff + 16);
        p0 = wmma_bf16(a0, bf, p0);
        p1 = wmma_bf16(a1, bf, p1);
      }
#pragma unroll
      for (int v = 0; v < 8; ++v) {
        int col = nt * 16 + ml;
        int r   = mt0 * 16 + v + 8 * hf;
        zpr[r * 128 + col]        = p0[v];
        zpr[(r + 16) * 128 + col] = p1[v];
      }
    }
    __syncthreads();

    // ---- epilogue: es / softmax(pi) / mu / exp(sigma) / tanh(rho) ----
    {
      int b = tid & 127, sec = tid >> 7;
      const float* zb = zpr + b * 128;
      size_t idx = (size_t)t * BATCH + b;
      size_t i20 = idx * 20;
      if (sec == 0) {
        out[idx] = 1.f / (1.f + __expf(zb[0] + be[0]));          // sigmoid(-z)
      } else if (sec == 1) {
        float zv[20], mx = -3.4e38f;
#pragma unroll
        for (int g2 = 0; g2 < 20; ++g2) { zv[g2] = zb[1 + g2] + bpi[g2]; mx = fmaxf(mx, zv[g2]); }
        float s = 0.f;
#pragma unroll
        for (int g2 = 0; g2 < 20; ++g2) { zv[g2] = __expf(zv[g2] - mx); s += zv[g2]; }
        float inv = 1.f / s;
#pragma unroll
        for (int g2 = 0; g2 < 20; ++g2) out[TBo + i20 + g2] = zv[g2] * inv;
      } else if (sec == 2) {
        for (int g2 = 0; g2 < 20; ++g2) out[TBo + NGo     + i20 + g2] = zb[21 + g2] + bm1[g2];
      } else if (sec == 3) {
        for (int g2 = 0; g2 < 20; ++g2) out[TBo + 2 * NGo + i20 + g2] = zb[41 + g2] + bm2[g2];
      } else if (sec == 4) {
        for (int g2 = 0; g2 < 20; ++g2) out[TBo + 3 * NGo + i20 + g2] = __expf(zb[61 + g2] + bs1[g2]);
      } else if (sec == 5) {
        for (int g2 = 0; g2 < 20; ++g2) out[TBo + 4 * NGo + i20 + g2] = __expf(zb[81 + g2] + bs2[g2]);
      } else if (sec == 6) {
        for (int g2 = 0; g2 < 20; ++g2) out[TBo + 5 * NGo + i20 + g2] = tanhf(zb[101 + g2] + br[g2]);
      }
    }
    __syncthreads();
  }
}

// ---------------------------------------------------------------------------
extern "C" void kernel_launch(void* const* d_in, const int* in_sizes, int n_in,
                              void* d_out, int out_size, void* d_ws, size_t ws_size,
                              hipStream_t stream) {
  (void)in_sizes; (void)n_in; (void)out_size; (void)ws_size;
  const float* x     = (const float*)d_in[0];
  const float* cmat  = (const float*)d_in[1];
  const float* W1_ih = (const float*)d_in[2];
  const float* W1_hh = (const float*)d_in[3];
  const float* b1    = (const float*)d_in[4];
  const float* W2_ih = (const float*)d_in[5];
  const float* W2_hh = (const float*)d_in[6];
  const float* b2    = (const float*)d_in[7];
  const float* W3_ih = (const float*)d_in[8];
  const float* W3_hh = (const float*)d_in[9];
  const float* b3    = (const float*)d_in[10];
  const float* Ww    = (const float*)d_in[11];
  const float* bw    = (const float*)d_in[12];
  const float* We    = (const float*)d_in[13];
  const float* be    = (const float*)d_in[14];
  const float* Wpi   = (const float*)d_in[15];
  const float* bpi   = (const float*)d_in[16];
  const float* Wm1   = (const float*)d_in[17];
  const float* bm1   = (const float*)d_in[18];
  const float* Wm2   = (const float*)d_in[19];
  const float* bm2   = (const float*)d_in[20];
  const float* Ws1   = (const float*)d_in[21];
  const float* bs1   = (const float*)d_in[22];
  const float* Ws2   = (const float*)d_in[23];
  const float* bs2   = (const float*)d_in[24];
  const float* Wr    = (const float*)d_in[25];
  const float* br    = (const float*)d_in[26];

  char* ws = (char*)d_ws;
  unsigned short* Wsw1 = (unsigned short*)(ws + WS_W1);
  unsigned short* Wsw2 = (unsigned short*)(ws + WS_W2);
  unsigned short* Wsw3 = (unsigned short*)(ws + WS_W3);
  unsigned short* Wswp = (unsigned short*)(ws + WS_WP);
  float* c1g = (float*)(ws + WS_C1);
  float* c2g = (float*)(ws + WS_C2);
  float* c3g = (float*)(ws + WS_C3);

  // weight repack: 1,638,400 elements total
  hs_prep_weights<<<6400, 256, 0, stream>>>(
      W1_ih, W1_hh, b1, W2_ih, W2_hh, b2, W3_ih, W3_hh, b3,
      We, Wpi, Wm1, Wm2, Ws1, Ws2, Wr, Wsw1, Wsw2, Wsw3, Wswp);

  (void)hipFuncSetAttribute(reinterpret_cast<const void*>(hs_recurrence),
                            hipFuncAttributeMaxDynamicSharedMemorySize, SMEM_BYTES);

  hs_recurrence<<<1, 1024, SMEM_BYTES, stream>>>(
      x, cmat, Ww, bw, be, bpi, bm1, bm2, bs1, bs2, br,
      (const char*)Wsw1, (const char*)Wsw2, (const char*)Wsw3, (const char*)Wswp,
      c1g, c2g, c3g, (float*)d_out);
}
